// STDP_87308095193752
// MI455X (gfx1250) — compile-verified, matrix operand order checked by reading, not run on hardware
//
#include <hip/hip_runtime.h>

typedef __attribute__((ext_vector_type(16))) _Float16 v16h;
typedef __attribute__((ext_vector_type(8)))  float    v8f;

constexpr int T_STEPS = 15;
constexpr int CIN     = 64;
constexpr int HDIM    = 256;
constexpr int WDIM    = 256;
constexpr int COUT    = 128;
constexpr int HO      = 252;
constexpr int WO      = 252;
constexpr int KWIN    = 16;
constexpr int PATCH   = CIN * 5 * 5;   // 1600

// ---------------------------------------------------------------------------
// Kernel 1: build cout->winner map and the 16 out_lat scalars.
// ---------------------------------------------------------------------------
__global__ void stdp_setup_kernel(const float* __restrict__ out_spikes,
                                  const int*   __restrict__ winners,
                                  float* __restrict__ out_vals,
                                  int*   __restrict__ cmap) {
    int tid = threadIdx.x;
    if (tid < COUT) {
        int k = -1;
        for (int i = 0; i < KWIN; ++i)
            if (winners[3 * i] == tid) k = i;
        cmap[tid] = k;
    }
    if (tid < KWIN) {
        int f = winners[3 * tid + 0];
        int r = winners[3 * tid + 1];
        int c = winners[3 * tid + 2];
        long long base = (long long)f * (HO * WO) + (long long)r * WO + c;
        float s = 0.0f;
        for (int t = 0; t < T_STEPS; ++t)
            s += out_spikes[(long long)t * (COUT * HO * WO) + base];
        out_vals[tid] = s;
    }
}

// ---------------------------------------------------------------------------
// Kernel 2: temporal patch sums via WMMA.
//   A (16x32 f16) = all ones, B (32x16 f16) = spike values:
//   column n = patch element (lane % 16), K-slots = timesteps
//   (lanes 0-15 carry t=0..7, lanes 16-31 carry t=8..14, rest zero).
//   D[m][n] = sum_k B[k][n]  -> exact temporal sum (spikes in {0,1}, <=15).
//   One wave produces 16 patch-element sums; 16 winners * 100 groups = 1600
//   waves = 200 blocks * 8 waves, full EXEC at the WMMA.
// ---------------------------------------------------------------------------
__global__ void stdp_patch_wmma_kernel(const float* __restrict__ in_spikes,
                                       const int*   __restrict__ winners,
                                       float* __restrict__ psum) {
    int wave = blockIdx.x * (blockDim.x >> 5) + (threadIdx.x >> 5);
    int lane = threadIdx.x & 31;
    int k = wave / 100;          // winner index 0..15
    int g = wave - k * 100;      // element-group 0..99
    int n = lane & 15;
    bool hi = lane >= 16;

    int r = winners[3 * k + 1];
    int c = winners[3 * k + 2];

    int e   = g * 16 + n;        // patch element 0..1599
    int ci  = e / 25;
    int rem = e - ci * 25;
    int kh  = rem / 5;
    int kw  = rem - kh * 5;
    long long addr0 = (long long)ci * (HDIM * WDIM)
                    + (long long)(r + kh) * WDIM + (c + kw);

    v16h a;
    for (int j = 0; j < 16; ++j) a[j] = (_Float16)1.0f;

    v16h b = {};
    int tbase = hi ? 8 : 0;
    for (int j = 0; j < 8; ++j) {
        int t = tbase + j;
        float s = 0.0f;
        if (t < T_STEPS)
            s = in_spikes[(long long)t * (CIN * HDIM * WDIM) + addr0];
        b[j] = (_Float16)s;
    }

    v8f acc = {};
    acc = __builtin_amdgcn_wmma_f32_16x16x32_f16(false, a, false, b,
                                                 (short)0, acc, false, false);

    // Row M=0 of column n lives in acc[0] on lanes 0-15.
    if (!hi) psum[k * PATCH + e] = acc[0];
}

// ---------------------------------------------------------------------------
// Kernel 3: elementwise STDP weight update + clip.
// ---------------------------------------------------------------------------
__global__ void stdp_update_kernel(const float* __restrict__ weight,
                                   const float* __restrict__ ltp,
                                   const float* __restrict__ ltd,
                                   const float* __restrict__ out_vals,
                                   const int*   __restrict__ cmap,
                                   const float* __restrict__ psum,
                                   float* __restrict__ out) {
    int idx = blockIdx.x * blockDim.x + threadIdx.x;
    if (idx >= COUT * PATCH) return;
    int co  = idx / PATCH;
    int rem = idx - co * PATCH;

    float w  = weight[idx];
    float nw = w;
    int k = cmap[co];
    if (k >= 0) {
        float lr = (psum[k * PATCH + rem] >= out_vals[k]) ? ltp[co] : ltd[co];
        nw = w + lr * (w * (1.0f - w));   // stab = (w-0)*(1-w)
    }
    nw = fminf(fmaxf(nw, 0.0f), 1.0f);
    out[idx] = nw;
}

// ---------------------------------------------------------------------------
extern "C" void kernel_launch(void* const* d_in, const int* in_sizes, int n_in,
                              void* d_out, int out_size, void* d_ws, size_t ws_size,
                              hipStream_t stream) {
    const float* in_spikes  = (const float*)d_in[0]; // (15,64,256,256)
    const float* out_spikes = (const float*)d_in[1]; // (15,128,252,252)
    const int*   winners    = (const int*)  d_in[2]; // (16,3)
    const float* weight     = (const float*)d_in[3]; // (128,64,5,5)
    const float* ltp        = (const float*)d_in[4]; // (128,)
    const float* ltd        = (const float*)d_in[5]; // (128,)
    float* out = (float*)d_out;                      // (128,64,5,5)

    // Workspace layout (all regions fully written before read):
    float* out_vals = (float*)d_ws;                        // 16 f32
    int*   cmap     = (int*)  ((char*)d_ws + 256);         // 128 i32
    float* psum     = (float*)((char*)d_ws + 1024);        // 16*1600 f32

    stdp_setup_kernel<<<1, 128, 0, stream>>>(out_spikes, winners, out_vals, cmap);

    // 16 winners * 100 groups = 1600 waves -> 200 blocks of 8 waves.
    stdp_patch_wmma_kernel<<<200, 256, 0, stream>>>(in_spikes, winners, psum);

    int total = COUT * PATCH;
    stdp_update_kernel<<<(total + 255) / 256, 256, 0, stream>>>(
        weight, ltp, ltd, out_vals, cmap, psum, out);
}
